// RWKV_Block_53223234732573
// MI455X (gfx1250) — matrix operand order repaired and checked by reading, not compiled, not run on hardware
//
#include <hip/hip_runtime.h>
#include <hip/hip_bf16.h>
#include <cstdint>
#include <cstddef>

// ---------------------------------------------------------------------------
// RWKV-6 block forward for MI455X (gfx1250, wave32, WMMA).
// All big GEMMs run as bf16 WMMA (v_wmma_f32_16x16x32_bf16) with f32
// accumulate; every pointwise op is fused into a GEMM epilogue.
// Weights are converted once per launch to bf16 [N][K] (transposed) so that
// both A and B WMMA fragments are contiguous 16B rows (ds_load_b128 friendly).
// bf16 weights total ~104 MB => resident in the 192 MB global L2.
// GEMM: 128x128 block tile, 8 waves x (32x64) wave tile => 8 WMMA / K-step
// per wave, double-buffered LDS, unconditional (padded) tile loads.
// ---------------------------------------------------------------------------

typedef __bf16 bf16_t;
typedef __attribute__((ext_vector_type(16))) bf16_t v16bf;
typedef __attribute__((ext_vector_type(8)))  bf16_t v8bf;
typedef __attribute__((ext_vector_type(8)))  float  v8f;

#define LN_EPS   1e-5f
#define GN_EPS_C 0.00064f

// ---------------- weight convert + transpose: f32 [R][Cn] -> bf16 [Cn][R] ---
__global__ __launch_bounds__(256) void k_transpose_cvt(
    const float* __restrict__ in, bf16_t* __restrict__ out, int R, int Cn)
{
  __shared__ float tile[32][33];
  const int c0 = blockIdx.x * 32, r0 = blockIdx.y * 32;
  const int tx = threadIdx.x, ty = threadIdx.y;   // blockDim = (32, 8)
#pragma unroll
  for (int i = 0; i < 32; i += 8) {
    int r = r0 + ty + i, c = c0 + tx;
    tile[ty + i][tx] = (r < R && c < Cn) ? in[(size_t)r * Cn + c] : 0.0f;
  }
  __syncthreads();
#pragma unroll
  for (int i = 0; i < 32; i += 8) {
    int c = c0 + ty + i, r = r0 + tx;
    if (c < Cn && r < R) out[(size_t)c * R + r] = (bf16_t)tile[tx][ty + i];
  }
}

// ---------------- LayerNorm over C (one block per row) ----------------------
__global__ __launch_bounds__(256) void k_layernorm(
    const float* __restrict__ x, const float* __restrict__ w,
    const float* __restrict__ b, float* __restrict__ out, int Cn)
{
  __shared__ float s1[256];
  __shared__ float s2[256];
  const size_t row = blockIdx.x;
  const float* xr = x + row * Cn;
  float sum = 0.f, sq = 0.f;
  for (int c = threadIdx.x; c < Cn; c += 256) { float v = xr[c]; sum += v; sq += v * v; }
  s1[threadIdx.x] = sum; s2[threadIdx.x] = sq;
  __syncthreads();
  for (int st = 128; st > 0; st >>= 1) {
    if ((int)threadIdx.x < st) {
      s1[threadIdx.x] += s1[threadIdx.x + st];
      s2[threadIdx.x] += s2[threadIdx.x + st];
    }
    __syncthreads();
  }
  const float mu  = s1[0] / Cn;
  const float var = s2[0] / Cn - mu * mu;
  const float inv = rsqrtf(var + LN_EPS);
  float* orow = out + row * Cn;
  for (int c = threadIdx.x; c < Cn; c += 256) orow[c] = (xr[c] - mu) * inv * w[c] + b[c];
}

// ---------------- token-shift mix -> bf16: out = xa + (x[t-1]-xa)*maa -------
__global__ __launch_bounds__(256) void k_mix_bf16(
    const float* __restrict__ xa, const float* __restrict__ shift,
    const float* __restrict__ maa, bf16_t* __restrict__ out,
    int T, int Cn, long total)
{
  long idx = (long)blockIdx.x * 256 + threadIdx.x;
  if (idx >= total) return;
  const int  c   = (int)(idx % Cn);
  const long row = idx / Cn;
  const int  t   = (int)(row % T);
  const float xav  = xa[idx];
  const float prev = (t == 0) ? shift[(row / T) * Cn + c] : xa[idx - Cn];
  out[idx] = (bf16_t)(xav + (prev - xav) * maa[c]);
}

// ---------------- copy last timestep row (new shift state) ------------------
__global__ __launch_bounds__(256) void k_copy_last(
    const float* __restrict__ xa, float* __restrict__ dst, int T, int Cn, int total)
{
  int idx = blockIdx.x * 256 + threadIdx.x;
  if (idx >= total) return;
  int b = idx / Cn, c = idx % Cn;
  dst[idx] = xa[((size_t)b * T + (T - 1)) * Cn + c];
}

// ---------------- GroupNorm over HS=64 + gate by g, emit bf16 ---------------
__global__ __launch_bounds__(256) void k_groupnorm_gate(
    const float* __restrict__ att, const float* __restrict__ g,
    const float* __restrict__ lw, const float* __restrict__ lb,
    bf16_t* __restrict__ gg, int Cn)
{
  const size_t row = blockIdx.x;
  const int lane = threadIdx.x & 31;
  const int wv   = threadIdx.x >> 5;              // 8 waves, 4 heads each
#pragma unroll
  for (int it = 0; it < 4; ++it) {
    const int h = wv * 4 + it;
    const size_t base = row * Cn + (size_t)h * 64;
    float a0 = att[base + lane], a1 = att[base + 32 + lane];
    float s = a0 + a1, q = a0 * a0 + a1 * a1;
#pragma unroll
    for (int m = 16; m > 0; m >>= 1) { s += __shfl_xor(s, m, 32); q += __shfl_xor(q, m, 32); }
    const float mu  = s * (1.0f / 64.0f);
    const float var = q * (1.0f / 64.0f) - mu * mu;
    const float inv = rsqrtf(var + GN_EPS_C);
    const int c0 = h * 64 + lane, c1 = c0 + 32;
    const float gn0 = (a0 - mu) * inv * lw[c0] + lb[c0];
    const float gn1 = (a1 - mu) * inv * lw[c1] + lb[c1];
    gg[row * Cn + c0] = (bf16_t)(gn0 * g[row * Cn + c0]);
    gg[row * Cn + c1] = (bf16_t)(gn1 * g[row * Cn + c1]);
  }
}

// ---------------- WKV-6 recurrent scan: one block per (b,h) -----------------
__global__ __launch_bounds__(256) void k_wkv_scan(
    const float* __restrict__ r, const float* __restrict__ k,
    const float* __restrict__ v, const float* __restrict__ w,
    const float* __restrict__ u, const float* __restrict__ s0,
    float* __restrict__ out, float* __restrict__ sout, int T, int H)
{
  const int HS = 64;
  const int bh = blockIdx.x;
  const int b = bh / H, h = bh % H;
  const int tid = threadIdx.x;
  const int j  = tid & 63;
  const int ig = tid >> 6;

  __shared__ float sr[64];
  __shared__ float sk[64];
  __shared__ float sv[64];
  __shared__ float sw[64];
  __shared__ float red[256];

  float S[16], uu[16];
#pragma unroll
  for (int ii = 0; ii < 16; ++ii) {
    const int i = ig * 16 + ii;
    uu[ii] = u[h * HS + i];
    S[ii]  = s0[(((size_t)b * H + h) * HS + i) * HS + j];
  }

  const int Cn = H * HS;
  for (int t = 0; t < T; ++t) {
    const size_t base = ((size_t)b * T + t) * Cn + (size_t)h * HS;
    if      (tid < 64)  sr[tid]       = r[base + tid];
    else if (tid < 128) sk[tid - 64]  = k[base + tid - 64];
    else if (tid < 192) sv[tid - 128] = v[base + tid - 128];
    else                sw[tid - 192] = w[base + tid - 192];
    __syncthreads();

    const float vj = sv[j];
    float acc = 0.f;
#pragma unroll
    for (int ii = 0; ii < 16; ++ii) {
      const int i = ig * 16 + ii;
      const float kv = sk[i] * vj;
      acc  += sr[i] * (S[ii] + uu[ii] * kv);
      S[ii] = sw[i] * S[ii] + kv;
    }
    red[tid] = acc;
    __syncthreads();
    if (tid < 64)
      out[base + tid] = red[tid] + red[64 + tid] + red[128 + tid] + red[192 + tid];
  }
#pragma unroll
  for (int ii = 0; ii < 16; ++ii) {
    const int i = ig * 16 + ii;
    sout[(((size_t)b * H + h) * HS + i) * HS + j] = S[ii];
  }
}

// ---------------- fused-epilogue bf16 WMMA GEMM -----------------------------
enum : int { EPI_F32 = 0, EPI_TANH, EPI_MIX, EPI_SILU, EPI_WDEC,
             EPI_RELUSQ, EPI_ADDX, EPI_SIG, EPI_FINAL };

struct EpiParams {
  float*       outf;
  bf16_t*      outb;
  const float* a0;
  const float* a1;
  const float* a2;
  int ldo;
  int T;
};

template <int EPI>
__device__ __forceinline__ void epi_write(float acc, int row, int col, const EpiParams& ep)
{
  const size_t idx = (size_t)row * ep.ldo + col;
  if constexpr (EPI == EPI_F32) {
    ep.outf[idx] = acc;
  } else if constexpr (EPI == EPI_TANH) {
    ep.outb[idx] = (bf16_t)tanhf(acc);
  } else if constexpr (EPI == EPI_MIX) {
    // out = xa + (x_prev - xa) * (maa + acc)
    const int t = row % ep.T;
    const int b = row / ep.T;
    const float xav   = ep.a0[idx];
    const float xprev = (t == 0) ? ep.a1[(size_t)b * ep.ldo + col] : ep.a0[idx - ep.ldo];
    ep.outb[idx] = (bf16_t)(xav + (xprev - xav) * (ep.a2[col] + acc));
  } else if constexpr (EPI == EPI_SILU) {
    ep.outf[idx] = acc / (1.0f + __expf(-acc));
  } else if constexpr (EPI == EPI_WDEC) {
    ep.outf[idx] = __expf(-__expf(ep.a0[col] + acc));
  } else if constexpr (EPI == EPI_RELUSQ) {
    const float rp = fmaxf(acc, 0.0f);
    ep.outb[idx] = (bf16_t)(rp * rp);
  } else if constexpr (EPI == EPI_ADDX) {
    ep.outf[idx] = ep.a0[idx] + acc;
  } else if constexpr (EPI == EPI_SIG) {
    ep.outf[idx] = 1.0f / (1.0f + __expf(-acc));
  } else if constexpr (EPI == EPI_FINAL) {
    ep.outf[idx] = ep.a0[idx] + ep.a1[idx] * acc;   // x2 = x1 + rr * acc
  }
}

// D = A[M][K](lda) * BT[N][K](ldb)^T, bf16 in / f32 accumulate.
// Block = 256 threads = 8 waves; block tile 128x128; wave tile 32x64
// (2 M-subtiles x 4 N-subtiles = 8 WMMAs per K-step per wave).
// K-loop step 32, double-buffered LDS (32 KB). Tile loads are UNCONDITIONAL:
// caller guarantees M % 128 == 0 and that BT has at least ceil(N/128)*128
// allocated rows (outputs beyond N are computed but never stored).
template <int EPI>
__global__ __launch_bounds__(256) void k_gemm_bf16_wmma(
    const bf16_t* __restrict__ A, int lda,
    const bf16_t* __restrict__ BT, int ldb,
    int M, int N, int K, EpiParams ep)
{
  __shared__ alignas(16) bf16_t As[2][128][32];
  __shared__ alignas(16) bf16_t Bs[2][128][32];

  const int tid  = threadIdx.x;
  const int lrow = tid >> 1;          // 0..127 : LDS tile row loaded by thread
  const int lhalf = tid & 1;          // 0..1   : 16-element half of the 32-K row
  const int blockM = blockIdx.y * 128;
  const int blockN = blockIdx.x * 128;

  const int wave = tid >> 5;          // 0..7
  const int lane = tid & 31;
  const int wm = wave & 3;            // wave M group (0..3) -> 32 rows
  const int wn = wave >> 2;           // wave N group (0..1) -> 64 cols
  const int lr = lane & 15;
  const int kh = lane >> 4;

  v8f acc[2][4];
#pragma unroll
  for (int mi = 0; mi < 2; ++mi)
#pragma unroll
    for (int ni = 0; ni < 4; ++ni) acc[mi][ni] = v8f{};

  const int nk = K >> 5;
  const bf16_t* aptr = A  + (size_t)(blockM + lrow) * lda + lhalf * 16;
  const bf16_t* bptr = BT + (size_t)(blockN + lrow) * ldb + lhalf * 16;

  v8bf ra0 = *(const v8bf*)(aptr);
  v8bf ra1 = *(const v8bf*)(aptr + 8);
  v8bf rb0 = *(const v8bf*)(bptr);
  v8bf rb1 = *(const v8bf*)(bptr + 8);

  int cur = 0;
  for (int kt = 0; kt < nk; ++kt) {
    __syncthreads();
    *(v8bf*)(&As[cur][lrow][lhalf * 16])     = ra0;
    *(v8bf*)(&As[cur][lrow][lhalf * 16 + 8]) = ra1;
    *(v8bf*)(&Bs[cur][lrow][lhalf * 16])     = rb0;
    *(v8bf*)(&Bs[cur][lrow][lhalf * 16 + 8]) = rb1;
    __syncthreads();

    if (kt + 1 < nk) {
      const int k0 = (kt + 1) << 5;
      ra0 = *(const v8bf*)(aptr + k0);
      ra1 = *(const v8bf*)(aptr + k0 + 8);
      rb0 = *(const v8bf*)(bptr + k0);
      rb1 = *(const v8bf*)(bptr + k0 + 8);
      if (kt + 2 < nk) {
        const int k1 = (kt + 2) << 5;
        __builtin_prefetch(aptr + k1, 0, 0);
        __builtin_prefetch(bptr + k1, 0, 0);
      }
    }

    // A fragments (16x32): lane = (m, kh); halfwords 0-7 -> K = 8*kh + h,
    // halfwords 8-15 -> K = 16 + 8*kh + (h-8)   [ISA 16-bit A layout]
    v16bf af[2];
#pragma unroll
    for (int mi = 0; mi < 2; ++mi) {
      const bf16_t* ap = &As[cur][wm * 32 + mi * 16 + lr][0];
      v8bf lo = *(const v8bf*)(ap + kh * 8);
      v8bf hi = *(const v8bf*)(ap + 16 + kh * 8);
      af[mi] = __builtin_shufflevector(lo, hi,
          0, 1, 2, 3, 4, 5, 6, 7, 8, 9, 10, 11, 12, 13, 14, 15);
    }
    // B fragments (32x16): lane = (n, kh); halfwords 0-7 -> K = 8*kh + h,
    // halfwords 8-15 -> K = 16 + 8*kh + (h-8)   (same as A)
    v16bf bfr[4];
#pragma unroll
    for (int ni = 0; ni < 4; ++ni) {
      const bf16_t* bp = &Bs[cur][wn * 64 + ni * 16 + lr][0];
      v8bf lo = *(const v8bf*)(bp + kh * 8);
      v8bf hi = *(const v8bf*)(bp + 16 + kh * 8);
      bfr[ni] = __builtin_shufflevector(lo, hi,
          0, 1, 2, 3, 4, 5, 6, 7, 8, 9, 10, 11, 12, 13, 14, 15);
    }

#pragma unroll
    for (int mi = 0; mi < 2; ++mi)
#pragma unroll
      for (int ni = 0; ni < 4; ++ni)
        acc[mi][ni] = __builtin_amdgcn_wmma_f32_16x16x32_bf16(
            false, af[mi], false, bfr[ni], (short)0, acc[mi][ni], false, false);

    cur ^= 1;
  }

  // D layout: VGPR v -> row m = v + 8*kh; col n = lane & 15.
#pragma unroll
  for (int mi = 0; mi < 2; ++mi) {
    const int mbase = blockM + wm * 32 + mi * 16 + kh * 8;
#pragma unroll
    for (int ni = 0; ni < 4; ++ni) {
      const int ncol = blockN + wn * 64 + ni * 16 + lr;
      if (ncol < N) {
#pragma unroll
        for (int vv = 0; vv < 8; ++vv) {
          const int mrow = mbase + vv;
          if (mrow < M) epi_write<EPI>(acc[mi][ni][vv], mrow, ncol, ep);
        }
      }
    }
  }
}

// ---------------------------------------------------------------------------
extern "C" void kernel_launch(void* const* d_in, const int* in_sizes, int n_in,
                              void* d_out, int out_size, void* d_ws, size_t ws_size,
                              hipStream_t stream)
{
  (void)in_sizes; (void)n_in; (void)out_size; (void)ws_size;
  constexpr int  Bb = 8, Tt = 2048, Cc = 2048, Hh = 32, FF = 7168;
  constexpr int  Mm = Bb * Tt;                 // 16384 rows
  constexpr long MC = (long)Mm * Cc;
  const int gy = Mm / 128;                     // GEMM grid.y

  const float* x       = (const float*)d_in[0];
  const float* att_sh  = (const float*)d_in[1];
  const float* ffn_sh  = (const float*)d_in[2];
  const float* wkv0    = (const float*)d_in[3];
  const float* ln1_w   = (const float*)d_in[4];
  const float* ln1_b   = (const float*)d_in[5];
  const float* ln2_w   = (const float*)d_in[6];
  const float* ln2_b   = (const float*)d_in[7];
  const float* maa_x   = (const float*)d_in[8];
  const float* maa[5]  = { (const float*)d_in[9],  (const float*)d_in[10],
                           (const float*)d_in[11], (const float*)d_in[12],
                           (const float*)d_in[13] };
  const float* tm_w1   = (const float*)d_in[14];
  const float* tm_w2   = (const float*)d_in[15];
  const float* tdecay  = (const float*)d_in[16];
  const float* td_w1   = (const float*)d_in[17];
  const float* td_w2   = (const float*)d_in[18];
  const float* tfaaaa  = (const float*)d_in[19];
  const float* Wr      = (const float*)d_in[20];
  const float* Wk      = (const float*)d_in[21];
  const float* Wv      = (const float*)d_in[22];
  const float* Wg      = (const float*)d_in[23];
  const float* Wo      = (const float*)d_in[24];
  const float* lnx_w   = (const float*)d_in[25];
  const float* lnx_b   = (const float*)d_in[26];
  const float* fmaa_k  = (const float*)d_in[27];
  const float* fmaa_r  = (const float*)d_in[28];
  const float* fWk     = (const float*)d_in[29];
  const float* fWv     = (const float*)d_in[30];
  const float* fWr     = (const float*)d_in[31];

  float* x2      = (float*)d_out;
  float* att_new = x2 + MC;
  float* ffn_new = att_new + (size_t)Bb * Cc;
  float* wkv_new = ffn_new + (size_t)Bb * Cc;

  char* ws = (char*)d_ws;
  size_t off = 0;
  auto take = [&](size_t bytes) -> void* {
    void* p = ws + off;
    off += (bytes + 255) & ~(size_t)255;
    return p;
  };

  // bf16 transposed weights (~104 MB total -> L2 resident).
  // NOTE: small-N weights are PADDED to a multiple of 128 rows so the GEMM
  // can load B tiles unconditionally (padded rows never reach the epilogue).
  bf16_t* WrT   = (bf16_t*)take((size_t)Cc * Cc * 2);
  bf16_t* WkT   = (bf16_t*)take((size_t)Cc * Cc * 2);
  bf16_t* WvT   = (bf16_t*)take((size_t)Cc * Cc * 2);
  bf16_t* WgT   = (bf16_t*)take((size_t)Cc * Cc * 2);
  bf16_t* WoT   = (bf16_t*)take((size_t)Cc * Cc * 2);
  bf16_t* fWrT  = (bf16_t*)take((size_t)Cc * Cc * 2);
  bf16_t* fWkT  = (bf16_t*)take((size_t)FF * Cc * 2);
  bf16_t* fWvT  = (bf16_t*)take((size_t)Cc * FF * 2);
  bf16_t* tmw1T = (bf16_t*)take((size_t)256 * Cc * 2);     // 160 -> pad 256
  bf16_t* tmw2T = (bf16_t*)take((size_t)5 * Cc * 32 * 2);
  bf16_t* tdw1T = (bf16_t*)take((size_t)128 * Cc * 2);     // 64 -> pad 128
  bf16_t* tdw2T = (bf16_t*)take((size_t)Cc * 64 * 2);

  // activations
  float*  xa    = (float*)take(MC * 4);
  bf16_t* dmix  = (bf16_t*)take(MC * 2);
  bf16_t* k5    = (bf16_t*)take((size_t)Mm * 160 * 2);
  bf16_t* x5[5];
  for (int f = 0; f < 5; ++f) x5[f] = (bf16_t*)take(MC * 2);  // xw,xk,xv,xr,xg
  float*  rbuf  = (float*)take(MC * 4);
  float*  kbuf  = (float*)take(MC * 4);
  float*  vbuf  = (float*)take(MC * 4);
  float*  gbuf  = (float*)take(MC * 4);
  bf16_t* wt64  = (bf16_t*)take((size_t)Mm * 64 * 2);
  float*  wdec  = (float*)take(MC * 4);
  float*  atto  = (float*)take(MC * 4);
  bf16_t* gg    = (bf16_t*)take(MC * 2);
  float*  x1    = (float*)take(MC * 4);
  float*  xf    = (float*)take(MC * 4);
  bf16_t* xkf   = (bf16_t*)take(MC * 2);
  bf16_t* xrf   = (bf16_t*)take(MC * 2);
  bf16_t* kk    = (bf16_t*)take((size_t)Mm * FF * 2);
  float*  rr    = (float*)take(MC * 4);

  // ---- phase 0: weight convert/transpose (f32 [K][N] -> bf16 [N][K]) ------
  const dim3 tb(32, 8);
  auto tgrid = [](int R, int Cn) { return dim3((Cn + 31) / 32, (R + 31) / 32); };
  k_transpose_cvt<<<tgrid(2048, 2048), tb, 0, stream>>>(Wr,  WrT,  2048, 2048);
  k_transpose_cvt<<<tgrid(2048, 2048), tb, 0, stream>>>(Wk,  WkT,  2048, 2048);
  k_transpose_cvt<<<tgrid(2048, 2048), tb, 0, stream>>>(Wv,  WvT,  2048, 2048);
  k_transpose_cvt<<<tgrid(2048, 2048), tb, 0, stream>>>(Wg,  WgT,  2048, 2048);
  k_transpose_cvt<<<tgrid(2048, 2048), tb, 0, stream>>>(Wo,  WoT,  2048, 2048);
  k_transpose_cvt<<<tgrid(2048, 2048), tb, 0, stream>>>(fWr, fWrT, 2048, 2048);
  k_transpose_cvt<<<tgrid(2048, FF),   tb, 0, stream>>>(fWk, fWkT, 2048, FF);
  k_transpose_cvt<<<tgrid(FF, 2048),   tb, 0, stream>>>(fWv, fWvT, FF, 2048);
  k_transpose_cvt<<<tgrid(2048, 160),  tb, 0, stream>>>(tm_w1, tmw1T, 2048, 160);
  for (int f = 0; f < 5; ++f)
    k_transpose_cvt<<<tgrid(32, 2048), tb, 0, stream>>>(
        tm_w2 + (size_t)f * 32 * Cc, tmw2T + (size_t)f * Cc * 32, 32, 2048);
  k_transpose_cvt<<<tgrid(2048, 64),   tb, 0, stream>>>(td_w1, tdw1T, 2048, 64);
  k_transpose_cvt<<<tgrid(64, 2048),   tb, 0, stream>>>(td_w2, tdw2T, 64, 2048);

  const int mixgrid = (int)((MC + 255) / 256);
  const int shgrid  = (Bb * Cc + 255) / 256;

  // ---- phase 1: ln1, time-mix LoRA ----------------------------------------
  k_layernorm<<<Mm, 256, 0, stream>>>(x, ln1_w, ln1_b, xa, Cc);
  k_mix_bf16<<<mixgrid, 256, 0, stream>>>(xa, att_sh, maa_x, dmix, Tt, Cc, MC);
  k_copy_last<<<shgrid, 256, 0, stream>>>(xa, att_new, Tt, Cc, Bb * Cc);

  // k5 = tanh(d @ tm_w1)  [K=2048, N=160 (padded B to 256 rows)]
  k_gemm_bf16_wmma<EPI_TANH><<<dim3(2, gy), 256, 0, stream>>>(
      dmix, Cc, tmw1T, Cc, Mm, 160, Cc,
      EpiParams{nullptr, k5, nullptr, nullptr, nullptr, 160, Tt});

  // x{w,k,v,r,g} = xa + dx*(maa_f + k5_f @ tm_w2_f)  [K=32, fused mix epilogue]
  for (int f = 0; f < 5; ++f)
    k_gemm_bf16_wmma<EPI_MIX><<<dim3(Cc / 128, gy), 256, 0, stream>>>(
        k5 + f * 32, 160, tmw2T + (size_t)f * Cc * 32, 32, Mm, Cc, 32,
        EpiParams{nullptr, x5[f], xa, att_sh, maa[f], Cc, Tt});

  // ---- phase 2: projections ------------------------------------------------
  k_gemm_bf16_wmma<EPI_F32><<<dim3(16, gy), 256, 0, stream>>>(
      x5[3], Cc, WrT, Cc, Mm, Cc, Cc,
      EpiParams{rbuf, nullptr, nullptr, nullptr, nullptr, Cc, Tt});
  k_gemm_bf16_wmma<EPI_F32><<<dim3(16, gy), 256, 0, stream>>>(
      x5[1], Cc, WkT, Cc, Mm, Cc, Cc,
      EpiParams{kbuf, nullptr, nullptr, nullptr, nullptr, Cc, Tt});
  k_gemm_bf16_wmma<EPI_F32><<<dim3(16, gy), 256, 0, stream>>>(
      x5[2], Cc, WvT, Cc, Mm, Cc, Cc,
      EpiParams{vbuf, nullptr, nullptr, nullptr, nullptr, Cc, Tt});
  k_gemm_bf16_wmma<EPI_SILU><<<dim3(16, gy), 256, 0, stream>>>(
      x5[4], Cc, WgT, Cc, Mm, Cc, Cc,
      EpiParams{gbuf, nullptr, nullptr, nullptr, nullptr, Cc, Tt});
  // decay LoRA: wt = tanh(xw @ td_w1); w = exp(-exp(decay + wt @ td_w2))
  k_gemm_bf16_wmma<EPI_TANH><<<dim3(1, gy), 256, 0, stream>>>(
      x5[0], Cc, tdw1T, Cc, Mm, 64, Cc,
      EpiParams{nullptr, wt64, nullptr, nullptr, nullptr, 64, Tt});
  k_gemm_bf16_wmma<EPI_WDEC><<<dim3(16, gy), 256, 0, stream>>>(
      wt64, 64, tdw2T, 64, Mm, Cc, 64,
      EpiParams{wdec, nullptr, tdecay, nullptr, nullptr, Cc, Tt});

  // ---- phase 3: WKV scan + groupnorm/gate + output proj --------------------
  k_wkv_scan<<<Bb * Hh, 256, 0, stream>>>(rbuf, kbuf, vbuf, wdec, tfaaaa, wkv0,
                                          atto, wkv_new, Tt, Hh);
  k_groupnorm_gate<<<Mm, 256, 0, stream>>>(atto, gbuf, lnx_w, lnx_b, gg, Cc);
  k_gemm_bf16_wmma<EPI_ADDX><<<dim3(16, gy), 256, 0, stream>>>(
      gg, Cc, WoT, Cc, Mm, Cc, Cc,
      EpiParams{x1, nullptr, x, nullptr, nullptr, Cc, Tt});

  // ---- phase 4: FFN --------------------------------------------------------
  k_layernorm<<<Mm, 256, 0, stream>>>(x1, ln2_w, ln2_b, xf, Cc);
  k_copy_last<<<shgrid, 256, 0, stream>>>(xf, ffn_new, Tt, Cc, Bb * Cc);
  k_mix_bf16<<<mixgrid, 256, 0, stream>>>(xf, ffn_sh, fmaa_k, xkf, Tt, Cc, MC);
  k_mix_bf16<<<mixgrid, 256, 0, stream>>>(xf, ffn_sh, fmaa_r, xrf, Tt, Cc, MC);

  k_gemm_bf16_wmma<EPI_RELUSQ><<<dim3(FF / 128, gy), 256, 0, stream>>>(
      xkf, Cc, fWkT, Cc, Mm, FF, Cc,
      EpiParams{nullptr, kk, nullptr, nullptr, nullptr, FF, Tt});
  k_gemm_bf16_wmma<EPI_SIG><<<dim3(16, gy), 256, 0, stream>>>(
      xrf, Cc, fWrT, Cc, Mm, Cc, Cc,
      EpiParams{rr, nullptr, nullptr, nullptr, nullptr, Cc, Tt});
  // x2 = x1 + rr * (kk @ fWv)   (written straight into d_out)
  k_gemm_bf16_wmma<EPI_FINAL><<<dim3(16, gy), 256, 0, stream>>>(
      kk, FF, fWvT, FF, Mm, Cc, FF,
      EpiParams{x2, nullptr, x1, rr, nullptr, Cc, Tt});
}
